// GRUDecoder_multistep_25623774888048
// MI455X (gfx1250) — matrix operand order repaired and checked by reading, not compiled, not run on hardware
//
#include <hip/hip_runtime.h>
#include <hip/hip_bf16.h>
#include <math.h>

typedef __attribute__((ext_vector_type(16))) _Float16 v16h;
typedef __attribute__((ext_vector_type(8)))  float    v8f;
typedef unsigned int u32x4 __attribute__((ext_vector_type(4)));
typedef int          i32x4 __attribute__((ext_vector_type(4)));
typedef int          i32x8 __attribute__((ext_vector_type(8)));

static constexpr int Bc = 8, Tc = 12, Nc = 100, Kc = 2;
static constexpr int GH = 64, MH = 64, MO = 64, NHd = 64;
static constexpr int Ec = Nc * (Nc - 1);          // 9900
static constexpr int W1H = Kc * 4 * 4 * 512;      // 16384 halfs (32 KB)
static constexpr int W2H = Kc * 2 * 4 * 512;      //  8192 halfs (16 KB)
static constexpr int WTOT = W1H + W2H;            // 24576 halfs (48 KB)

union AFrag { v16h v; float4 q[2]; };
union CFrag { v8f  v; float  f[8]; };

// ---- TDM path selection (reported to stderr via pragma message) ----
#if __has_builtin(__builtin_amdgcn_tensor_load_to_lds)
#  if __clang_major__ <= 22
#    define TDM_MODE 1
#    pragma message("TDM probe: using 5-arg __builtin_amdgcn_tensor_load_to_lds (clang<=22)")
#  else
#    define TDM_MODE 2
#    pragma message("TDM probe: using 6-arg __builtin_amdgcn_tensor_load_to_lds (clang>=23)")
#  endif
#else
#  define TDM_MODE 3
#  pragma message("TDM probe: builtin NOT available -> inline asm tensor_load_to_lds")
#endif

// wave-private LDS ordering: DS ops are in-order per wave; only stop the
// compiler from reordering across this point.
__device__ __forceinline__ void wave_sync() {
  asm volatile("" ::: "memory");
  __builtin_amdgcn_wave_barrier();
  asm volatile("" ::: "memory");
}

// branch-free transcendentals: v_exp_f32 + v_rcp_f32, no EXEC divergence.
__device__ __forceinline__ float fast_tanh(float x) {
  float e = __expf(x + x);
  return 1.f - 2.f * __builtin_amdgcn_rcpf(e + 1.f);
}
__device__ __forceinline__ float fast_sigmoid(float x) {
  return __builtin_amdgcn_rcpf(1.f + __expf(-x));
}

// -------------------------------------------------------------------------
__global__ void init_state(float* hidden, _Float16* hid16, float* prev) {
  int i = blockIdx.x * blockDim.x + threadIdx.x;
  if (i < Bc * Nc * GH) { hidden[i] = 0.f; hid16[i] = (_Float16)0.f; }
  if (i < Bc * Nc) prev[i] = 0.f;
}

// swizzle msg-MLP weights into WMMA B-fragment order (f16)
__global__ void prep_weights(const float* __restrict__ w1,
                             const float* __restrict__ w2,
                             _Float16* __restrict__ w1f,
                             _Float16* __restrict__ w2f) {
  int i = blockIdx.x * blockDim.x + threadIdx.x;
  if (i < W1H) {
    int e = i & 15, lane = (i >> 4) & 31, frag = i >> 9;
    int nt = frag & 3, kt = (frag >> 2) & 3, k = frag >> 4;
    int Krow = kt * 32 + ((lane >= 16) ? 8 : 0) + ((e < 8) ? e : e + 8);
    int col  = nt * 16 + (lane & 15);
    w1f[i] = (_Float16)w1[(k * 128 + Krow) * 64 + col];
  } else if (i < WTOT) {
    int j = i - W1H;
    int e = j & 15, lane = (j >> 4) & 31, frag = j >> 9;
    int nt = frag & 3, kt = (frag >> 2) & 1, k = frag >> 3;
    int Krow = kt * 32 + ((lane >= 16) ? 8 : 0) + ((e < 8) ? e : e + 8);
    int col  = nt * 16 + (lane & 15);
    w2f[j] = (_Float16)w2[(k * 64 + Krow) * 64 + col];
  }
}

// -------------------------------------------------------------------------
// edge message passing: one wave per (batch b, receiver j); 4 waves / block.
// 48 KB pre-swizzled weights staged once per block into LDS via TDM; the
// per-tile hidden-row gather is software-pipelined through registers.
// -------------------------------------------------------------------------
__global__ void __launch_bounds__(128)
edge_msgs(const _Float16* __restrict__ hid16,
          const float*    __restrict__ rel_types,
          const _Float16* __restrict__ wglob,   // w1f ‖ w2f, 49152 B contiguous
          const float*    __restrict__ b1,
          const float*    __restrict__ b2,
          float*          __restrict__ agg) {
  __shared__ _Float16 wlds[WTOT]        __attribute__((aligned(16)));
  __shared__ _Float16 preA[4][16][128]  __attribute__((aligned(16)));
  __shared__ _Float16 h1s [4][16][64]   __attribute__((aligned(16)));
  __shared__ float    rts [4][16][2];

  const int lane = threadIdx.x & 31;
  const int wv   = threadIdx.x >> 5;
  const int gw   = blockIdx.x * 4 + wv;        // 0..799
  const int b    = gw / Nc;
  const int j    = gw % Nc;                    // receiver node
  const int m15  = lane & 15;
  const int hi8  = (lane >= 16) ? 8 : 0;

  // ---- stage weights into LDS with the Tensor Data Mover ----
  if (wv == 0) {
    unsigned lds_off = (unsigned)(size_t)(&wlds[0]);
    unsigned long long ga = (unsigned long long)(size_t)wglob;
    u32x4 g0;
    g0[0] = 1u;                                            // count=1
    g0[1] = lds_off;                                       // lds_addr
    g0[2] = (unsigned)(ga & 0xFFFFFFFFull);                // global_addr lo
    g0[3] = (unsigned)((ga >> 32) & 0x1FFFFFFull) | (2u << 30);  // hi | type=2
    const unsigned td0 = (unsigned)(WTOT * 2 / 8);         // 6144 x 8B units
    i32x8 g1;
    g1[0] = (int)(3u << 16);                               // data_size=8B
    g1[1] = (int)((td0 & 0xFFFFu) << 16);                  // tensor_dim0 lo
    g1[2] = (int)(((td0 >> 16) & 0xFFFFu) | (1u << 16));   // dim0 hi | dim1=1
    g1[3] = (int)((td0 & 0xFFFFu) << 16);                  // tile_dim0
    g1[4] = 0;                                             // tile_dim1/2 unused
    g1[5] = (int)td0;                                      // dim0_stride lo
    g1[6] = 0;
    g1[7] = 0;
    i32x4 g2; g2[0] = 0; g2[1] = 0; g2[2] = 0; g2[3] = 0;
    i32x4 g3; g3[0] = 0; g3[1] = 0; g3[2] = 0; g3[3] = 0;
#if TDM_MODE == 1
    __builtin_amdgcn_tensor_load_to_lds(g0, g1, g2, g3, 0);
#elif TDM_MODE == 2
    i32x8 g5; g5[0]=0; g5[1]=0; g5[2]=0; g5[3]=0; g5[4]=0; g5[5]=0; g5[6]=0; g5[7]=0;
    __builtin_amdgcn_tensor_load_to_lds(g0, g1, g2, g3, g5, 0);
#else
    asm volatile("tensor_load_to_lds %0, %1, %2, %3"
                 :: "s"(g0), "s"(g1), "s"(g2), "s"(g3) : "memory");
#endif
    __builtin_amdgcn_s_wait_tensorcnt(0);
  }
  __syncthreads();
  const _Float16* w1l = &wlds[0];
  const _Float16* w2l = &wlds[W1H];

  // ---- hoist biases (depend only on k, nt, lane) ----
  float bias1[Kc][4], bias2[Kc][4];
#pragma unroll
  for (int k = 0; k < Kc; ++k)
#pragma unroll
    for (int nt = 0; nt < 4; ++nt) {
      bias1[k][nt] = b1[k * MH + nt * 16 + m15];
      bias2[k][nt] = b2[k * MO + nt * 16 + m15];
    }

  CFrag acc[4];
#pragma unroll
  for (int nt = 0; nt < 4; ++nt)
#pragma unroll
    for (int q = 0; q < 8; ++q) acc[nt].f[q] = 0.f;

  // ---- register-staged pipeline for the per-tile hidden gather ----
  float4 sreg[8]; float sr0, sr1;

#define STAGE_LOAD(TILE)                                                     \
  {                                                                          \
    const int s_ = (TILE) * 16 + m15;                                        \
    const bool valid_ = (s_ < Nc - 1);                                       \
    const int is_ = valid_ ? (s_ + (s_ >= j ? 1 : 0)) : 0;                   \
    const int row_ = (lane < 16) ? is_ : j;                                  \
    const float4* src_ = reinterpret_cast<const float4*>(                    \
        hid16 + ((size_t)b * Nc + row_) * GH);                               \
    _Pragma("unroll")                                                        \
    for (int q = 0; q < 8; ++q) sreg[q] = src_[q];                           \
    sr0 = 0.f; sr1 = 0.f;                                                    \
    if ((lane < 16) && valid_) {                                             \
      const int e_ = is_ * (Nc - 1) + ((j < is_) ? j : (j - 1));             \
      const float* rt_ = rel_types + ((size_t)b * Ec + e_) * Kc;             \
      sr0 = rt_[0]; sr1 = rt_[1];                                            \
    }                                                                        \
  }

#define STAGE_STORE()                                                        \
  {                                                                          \
    float4* dst_ = reinterpret_cast<float4*>(                                \
        &preA[wv][m15][(lane < 16) ? 0 : 64]);                               \
    _Pragma("unroll")                                                        \
    for (int q = 0; q < 8; ++q) dst_[q] = sreg[q];                           \
    if (lane < 16) { rts[wv][m15][0] = sr0; rts[wv][m15][1] = sr1; }         \
  }

  STAGE_LOAD(0)
  STAGE_STORE()
  wave_sync();

  for (int tile = 0; tile < 7; ++tile) {
    // ---- A fragments for layer 1 (last reads of preA this tile) ----
    AFrag aF[4];
#pragma unroll
    for (int kt = 0; kt < 4; ++kt) {
      const _Float16* p = &preA[wv][m15][kt * 32 + hi8];
      aF[kt].q[0] = *reinterpret_cast<const float4*>(p);
      aF[kt].q[1] = *reinterpret_cast<const float4*>(p + 16);
    }
    // issue next tile's global gather early; consumed after compute
    if (tile < 6) STAGE_LOAD(tile + 1)

#pragma unroll
    for (int k = 0; k < Kc; ++k) {
      // ----- layer 1: h1 = tanh(pre_msg @ W1_k + b1_k) -----
#pragma unroll
      for (int nt = 0; nt < 4; ++nt) {
        CFrag c;
#pragma unroll
        for (int q = 0; q < 8; ++q) c.f[q] = 0.f;
#pragma unroll
        for (int kt = 0; kt < 4; ++kt) {
          AFrag bF;
          const _Float16* p = w1l + ((size_t)((k * 4 + kt) * 4 + nt) * 32 + lane) * 16;
          bF.q[0] = *reinterpret_cast<const float4*>(p);
          bF.q[1] = *reinterpret_cast<const float4*>(p + 8);
          c.v = __builtin_amdgcn_wmma_f32_16x16x32_f16(
              false, aF[kt].v, false, bF.v, (short)0, c.v, false, false);
        }
#pragma unroll
        for (int q = 0; q < 8; ++q) {
          const float t = fast_tanh(c.f[q] + bias1[k][nt]);
          h1s[wv][q + hi8][nt * 16 + m15] = (_Float16)t;
        }
      }
      wave_sync();

      // ----- layer 2 -----
      AFrag a2F[2];
#pragma unroll
      for (int kt = 0; kt < 2; ++kt) {
        const _Float16* p = &h1s[wv][m15][kt * 32 + hi8];
        a2F[kt].q[0] = *reinterpret_cast<const float4*>(p);
        a2F[kt].q[1] = *reinterpret_cast<const float4*>(p + 16);
      }
#pragma unroll
      for (int nt = 0; nt < 4; ++nt) {
        CFrag c;
#pragma unroll
        for (int q = 0; q < 8; ++q) c.f[q] = 0.f;
#pragma unroll
        for (int kt = 0; kt < 2; ++kt) {
          AFrag bF;
          const _Float16* p = w2l + ((size_t)((k * 2 + kt) * 4 + nt) * 32 + lane) * 16;
          bF.q[0] = *reinterpret_cast<const float4*>(p);
          bF.q[1] = *reinterpret_cast<const float4*>(p + 8);
          c.v = __builtin_amdgcn_wmma_f32_16x16x32_f16(
              false, a2F[kt].v, false, bF.v, (short)0, c.v, false, false);
        }
#pragma unroll
        for (int q = 0; q < 8; ++q) {
          const float t = fast_tanh(c.f[q] + bias2[k][nt]);
          acc[nt].f[q] += t * rts[wv][q + hi8][k];
        }
      }
      wave_sync();     // h1s reused by next k
    }

    // overwrite preA/rts with the pre-gathered next tile
    if (tile < 6) STAGE_STORE()
    wave_sync();
  }
#undef STAGE_LOAD
#undef STAGE_STORE

  // ---- reduce over 16 edge rows of each C tile, write agg[b][j][:] ----
  const float scale = 1.0f / (float)(Kc * Nc);
#pragma unroll
  for (int nt = 0; nt < 4; ++nt) {
    float cs = 0.f;
#pragma unroll
    for (int q = 0; q < 8; ++q) cs += acc[nt].f[q];
    cs += __shfl_xor(cs, 16, 32);
    if (lane < 16)
      agg[((size_t)b * Nc + j) * MO + nt * 16 + lane] = cs * scale;
  }
}

// -------------------------------------------------------------------------
// per-node GRU cell + residual output MLP (fp32, 64 threads = 64 channels)
// -------------------------------------------------------------------------
__global__ void __launch_bounds__(64)
node_update(const float* __restrict__ x,
            const float* __restrict__ agg,
            float* __restrict__ hidden, _Float16* __restrict__ hid16,
            float* __restrict__ prev,
            const float* __restrict__ hrw, const float* __restrict__ hiw,
            const float* __restrict__ hnw,
            const float* __restrict__ irw, const float* __restrict__ irb,
            const float* __restrict__ iiw, const float* __restrict__ iib,
            const float* __restrict__ inw, const float* __restrict__ inb,
            const float* __restrict__ f1w, const float* __restrict__ f1b,
            const float* __restrict__ f2w, const float* __restrict__ f2b,
            const float* __restrict__ f3w, const float* __restrict__ f3b,
            const int* __restrict__ burn_in_steps,
            float* __restrict__ out, int t) {
  __shared__ float sAgg[GH], sHid[GH], sNew[GH], sP[NHd];
  const int b = blockIdx.x / Nc, n = blockIdx.x % Nc, c = threadIdx.x;
  const size_t base = ((size_t)b * Nc + n) * GH;
  sAgg[c] = agg[base + c];
  sHid[c] = hidden[base + c];
  const float ins = (t <= burn_in_steps[0]) ? x[((size_t)b * Tc + t) * Nc + n]
                                            : prev[b * Nc + n];
  __syncthreads();

  float accr = 0.f, acci = 0.f, accn = 0.f;
#pragma unroll 8
  for (int m = 0; m < GH; ++m) {
    const float a = sAgg[m];
    accr += a * hrw[m * GH + c];
    acci += a * hiw[m * GH + c];
    accn += a * hnw[m * GH + c];
  }
  const float r  = fast_sigmoid(ins * irw[c] + irb[c] + accr);
  const float ii = fast_sigmoid(ins * iiw[c] + iib[c] + acci);
  const float nn = fast_tanh(ins * inw[c] + inb[c] + r * accn);
  const float h  = (1.f - ii) * nn + ii * sHid[c];
  sNew[c] = h;
  hidden[base + c] = h;
  hid16[base + c] = (_Float16)h;
  __syncthreads();

  float p1 = f1b[c];
#pragma unroll 8
  for (int m = 0; m < NHd; ++m) p1 += sNew[m] * f1w[m * NHd + c];
  p1 = fmaxf(p1, 0.f);
  sP[c] = p1;
  __syncthreads();

  float p2 = f2b[c];
#pragma unroll 8
  for (int m = 0; m < NHd; ++m) p2 += sP[m] * f2w[m * NHd + c];
  p2 = fmaxf(p2, 0.f);
  __syncthreads();
  sP[c] = p2;
  __syncthreads();

  if (c == 0) {
    float pr = f3b[0];
    for (int m = 0; m < NHd; ++m) pr += sP[m] * f3w[m];
    pr += ins;
    out[((size_t)b * (Tc - 1) + t) * Nc + n] = pr;
    prev[b * Nc + n] = pr;
  }
}

// -------------------------------------------------------------------------
extern "C" void kernel_launch(void* const* d_in, const int* in_sizes, int n_in,
                              void* d_out, int out_size, void* d_ws, size_t ws_size,
                              hipStream_t stream) {
  const float* x         = (const float*)d_in[0];
  const float* rel_types = (const float*)d_in[3];
  const float* w1  = (const float*)d_in[4];
  const float* b1  = (const float*)d_in[5];
  const float* w2  = (const float*)d_in[6];
  const float* b2  = (const float*)d_in[7];
  const float* hrw = (const float*)d_in[8];
  const float* hiw = (const float*)d_in[9];
  const float* hnw = (const float*)d_in[10];
  const float* irw = (const float*)d_in[11];
  const float* irb = (const float*)d_in[12];
  const float* iiw = (const float*)d_in[13];
  const float* iib = (const float*)d_in[14];
  const float* inw = (const float*)d_in[15];
  const float* inb = (const float*)d_in[16];
  const float* f1w = (const float*)d_in[17];
  const float* f1b = (const float*)d_in[18];
  const float* f2w = (const float*)d_in[19];
  const float* f2b = (const float*)d_in[20];
  const float* f3w = (const float*)d_in[21];
  const float* f3b = (const float*)d_in[22];
  const int*   bis = (const int*)d_in[24];
  float* out = (float*)d_out;

  auto au = [](size_t v) { return (v + 255) & ~(size_t)255; };
  char* p = (char*)d_ws;
  float*    hidden = (float*)p;    p += au(sizeof(float)    * Bc * Nc * GH);
  _Float16* hid16  = (_Float16*)p; p += au(sizeof(_Float16) * Bc * Nc * GH);
  float*    prev   = (float*)p;    p += au(sizeof(float)    * Bc * Nc);
  float*    agg    = (float*)p;    p += au(sizeof(float)    * Bc * Nc * MO);
  _Float16* w1f    = (_Float16*)p; p += au(sizeof(_Float16) * W1H);  // 32768 B
  _Float16* w2f    = (_Float16*)p; p += au(sizeof(_Float16) * W2H);  // contiguous

  init_state<<<(Bc * Nc * GH + 255) / 256, 256, 0, stream>>>(hidden, hid16, prev);
  prep_weights<<<(WTOT + 255) / 256, 256, 0, stream>>>(w1, w2, w1f, w2f);

  for (int t = 0; t < Tc - 1; ++t) {
    edge_msgs<<<(Bc * Nc) / 4, 128, 0, stream>>>(hid16, rel_types, w1f, b1, b2, agg);
    node_update<<<Bc * Nc, 64, 0, stream>>>(x, agg, hidden, hid16, prev,
        hrw, hiw, hnw, irw, irb, iiw, iib, inw, inb,
        f1w, f1b, f2w, f2b, f3w, f3b, bis, out, t);
  }
}
